// GraphConvNet_8564164788759
// MI455X (gfx1250) — matrix operand order repaired and checked by reading, not compiled, not run on hardware
//
#include <hip/hip_runtime.h>
#include <hip/hip_bf16.h>

typedef __attribute__((ext_vector_type(16))) _Float16 v16h;
typedef __attribute__((ext_vector_type(8)))  _Float16 v8h;
typedef __attribute__((ext_vector_type(4)))  _Float16 v4h;
typedef __attribute__((ext_vector_type(8)))  float    v8f;

#define HID 64
#define NPG 200   // nodes per graph (batch is repeat(arange(G), 200))

#if defined(__HIP_DEVICE_COMPILE__) && __has_builtin(__builtin_amdgcn_global_load_async_to_lds_b64) && __has_builtin(__builtin_amdgcn_s_wait_asynccnt)
#define USE_ASYNC_LDS 1
#else
#define USE_ASYNC_LDS 0
#endif

#if USE_ASYNC_LDS
typedef int async_b64_t __attribute__((vector_size(8)));           // v2i payload type
typedef __attribute__((address_space(1))) async_b64_t* gasync_p;   // global side
typedef __attribute__((address_space(3))) async_b64_t* lasync_p;   // LDS side
#endif

__device__ __forceinline__ v8f wmma_f16(v16h a, v16h b, v8f c) {
    // D = A(16x32 f16) x B(32x16 f16) + C(16x16 f32)
    return __builtin_amdgcn_wmma_f32_16x16x32_f16(
        false, a, false, b, (short)0, c, false, false);
}

__device__ __forceinline__ v16h cat8(v8h lo, v8h hi) {
    return __builtin_shufflevector(lo, hi, 0, 1, 2, 3, 4, 5, 6, 7,
                                   8, 9, 10, 11, 12, 13, 14, 15);
}

// ---- pack a (K x Ncols) f32 weight matrix into f16 WMMA B-fragments.
// dst layout: frag = ks*NT + nt; dst[((frag*32)+lane)*16 + j] = W[k][col]
//   k = ks*32 + (lane>>4)*8 + (j&7) + (j>>3)*16, col = nt*16 + (lane&15); zero-pad k >= K.
__global__ void pack_weights(const float* __restrict__ W, _Float16* __restrict__ dst,
                             int K, int Ncols, int KS, int NT) {
    int t = blockIdx.x * blockDim.x + threadIdx.x;
    int total = KS * NT * 32 * 16;
    if (t >= total) return;
    int j = t & 15;
    int lane = (t >> 4) & 31;
    int frag = t >> 9;
    int nt = frag % NT, ks = frag / NT;
    int k = ks * 32 + (lane >> 4) * 8 + (j & 7) + (j >> 3) * 16;
    int col = nt * 16 + (lane & 15);
    dst[t] = (k < K) ? (_Float16)W[(size_t)k * Ncols + col] : (_Float16)0.f;
}

// ---- f32 -> f16 mirror (4 elements / thread)
__global__ void f32_to_f16(const float* __restrict__ src, _Float16* __restrict__ dst, int n4) {
    int t = blockIdx.x * blockDim.x + threadIdx.x;
    if (t >= n4) return;
    float4 v = ((const float4*)src)[t];
    v4h hv = {(_Float16)v.x, (_Float16)v.y, (_Float16)v.z, (_Float16)v.w};
    ((v4h*)dst)[t] = hv;
}

// ---------------- embedding + projection: x = emb[x_idx] @ proj_W + proj_b
__global__ void embed_proj(const int* __restrict__ xidx, const float* __restrict__ emb,
                           const float* __restrict__ W, const float* __restrict__ b,
                           float* __restrict__ x, _Float16* __restrict__ xh, int N) {
    int t = blockIdx.x * blockDim.x + threadIdx.x;
    int n = t >> 6, c = t & 63;
    if (n >= N) return;
    int id = xidx[n];
    float s = b[c];
#pragma unroll
    for (int k = 0; k < 16; ++k) s += emb[id * 16 + k] * W[k * 64 + c];
    x[(size_t)n * 64 + c] = s;
    xh[(size_t)n * 64 + c] = (_Float16)s;
}

// ---------------- edge scatter: agg[dst] += x[src] * ew   (64 lanes per edge)
__global__ void edge_scatter(const int* __restrict__ src, const int* __restrict__ dst,
                             const float* __restrict__ ew, const float* __restrict__ x,
                             float* __restrict__ agg, int E) {
    int t = blockIdx.x * blockDim.x + threadIdx.x;
    int e = t >> 6, c = t & 63;
    if (e >= E) return;
    float v = x[(size_t)src[e] * 64 + c] * ew[e];
    atomicAdd(&agg[(size_t)dst[e] * 64 + c], v);
}

// ---------------- conv: y = agg @ relW + relB + x @ rootW   (one wave per 16 rows)
__global__ void __launch_bounds__(32)
conv_wmma(const _Float16* __restrict__ aggh, const _Float16* __restrict__ xh,
          const _Float16* __restrict__ pkrel, const float* __restrict__ relB,
          const _Float16* __restrict__ pkroot,
          float* __restrict__ yout, _Float16* __restrict__ yhout) {
    const int lane = threadIdx.x;
    const int mrow = lane & 15, half = lane >> 4;
    const int rowbase = blockIdx.x * 16;

    v8f acc[4];
#pragma unroll
    for (int nt = 0; nt < 4; ++nt) {
        float bv = relB[nt * 16 + mrow];
#pragma unroll
        for (int v = 0; v < 8; ++v) acc[nt][v] = bv;
    }

#pragma unroll
    for (int m = 0; m < 2; ++m) {
        const _Float16* Arow = (m == 0 ? aggh : xh) + (size_t)(rowbase + mrow) * 64;
        const v16h* pk = (const v16h*)(m == 0 ? pkrel : pkroot);
#pragma unroll
        for (int ks = 0; ks < 2; ++ks) {
            int kb = ks * 32 + half * 8;
            v8h lo = *(const v8h*)(Arow + kb);
            v8h hi = *(const v8h*)(Arow + kb + 16);
            v16h a = cat8(lo, hi);
#pragma unroll
            for (int nt = 0; nt < 4; ++nt)
                acc[nt] = wmma_f16(a, pk[(ks * 4 + nt) * 32 + lane], acc[nt]);
        }
    }
#pragma unroll
    for (int nt = 0; nt < 4; ++nt)
#pragma unroll
        for (int v = 0; v < 8; ++v) {
            size_t idx = (size_t)(rowbase + half * 8 + v) * 64 + nt * 16 + mrow;
            yout[idx] = acc[nt][v];
            yhout[idx] = (_Float16)acc[nt][v];
        }
}

// ---------------- demo MLP (tiny): out = in @ W + b,  W is (K,16)
__global__ void demo_mlp(const float* __restrict__ in, const float* __restrict__ W,
                         const float* __restrict__ b, float* __restrict__ out,
                         int K, int G) {
    int t = blockIdx.x * blockDim.x + threadIdx.x;
    int g = t >> 4, c = t & 15;
    if (g >= G) return;
    float s = b[c];
    for (int k = 0; k < K; ++k) s += in[g * K + k] * W[k * 16 + c];
    out[g * 16 + c] = s;
}

// ---------------- down: x = leaky_relu([x, demo[batch]] @ W1 + b1) @ W2 + b2
__global__ void __launch_bounds__(32)
down_wmma(const _Float16* __restrict__ yh, const float* __restrict__ demo,
          const int* __restrict__ batch, const _Float16* __restrict__ pkw1,
          const float* __restrict__ b1, const _Float16* __restrict__ pkw2,
          const float* __restrict__ b2, float* __restrict__ xout) {
    __shared__ _Float16 comb[16 * 96];   // [x(64) | demo(16) | 0(16)] per row
    __shared__ _Float16 hsh[16 * 128];   // leaky_relu(h) as f16
    const int lane = threadIdx.x;
    const int mrow = lane & 15, half = lane >> 4;
    const int rowbase = blockIdx.x * 16;

    // ---- stage x-part of combined (16 rows x 64 f16)
#if USE_ASYNC_LDS
#pragma unroll
    for (int it = 0; it < 8; ++it) {
        int row = it * 2 + half;
        const _Float16* g = yh + (size_t)(rowbase + row) * 64 + mrow * 4;
        _Float16* l = comb + row * 96 + mrow * 4;
        __builtin_amdgcn_global_load_async_to_lds_b64(
            (gasync_p)(void*)const_cast<_Float16*>(g),
            (lasync_p)(void*)l, 0, 0);
    }
    __builtin_amdgcn_s_wait_asynccnt(0);
#else
    for (int t = lane; t < 16 * 64; t += 32) {
        int r = t >> 6, c = t & 63;
        comb[r * 96 + c] = yh[(size_t)(rowbase + r) * 64 + c];
    }
#endif
    // demo part + zero pad (cols 64..95)
    for (int t = lane; t < 16 * 32; t += 32) {
        int r = t >> 5, c = t & 31;
        comb[r * 96 + 64 + c] =
            (c < 16) ? (_Float16)demo[batch[rowbase + r] * 16 + c] : (_Float16)0.f;
    }
    __syncthreads();

    // GEMM1: (16x80pad96) x (80x128)
    v8f acc1[8];
#pragma unroll
    for (int nt = 0; nt < 8; ++nt) {
        float bv = b1[nt * 16 + mrow];
#pragma unroll
        for (int v = 0; v < 8; ++v) acc1[nt][v] = bv;
    }
    const v16h* pk1 = (const v16h*)pkw1;
#pragma unroll
    for (int ks = 0; ks < 3; ++ks) {
        int kb = ks * 32 + half * 8;
        v8h lo = *(const v8h*)(comb + mrow * 96 + kb);
        v8h hi = *(const v8h*)(comb + mrow * 96 + kb + 16);
        v16h a = cat8(lo, hi);
#pragma unroll
        for (int nt = 0; nt < 8; ++nt)
            acc1[nt] = wmma_f16(a, pk1[(ks * 8 + nt) * 32 + lane], acc1[nt]);
    }
    // leaky_relu -> LDS (f16)
#pragma unroll
    for (int nt = 0; nt < 8; ++nt)
#pragma unroll
        for (int v = 0; v < 8; ++v) {
            float hv = acc1[nt][v];
            hsh[(half * 8 + v) * 128 + nt * 16 + mrow] =
                (_Float16)(hv > 0.f ? hv : 0.01f * hv);
        }
    __syncthreads();

    // GEMM2: (16x128) x (128x64)
    v8f acc2[4];
#pragma unroll
    for (int nt = 0; nt < 4; ++nt) {
        float bv = b2[nt * 16 + mrow];
#pragma unroll
        for (int v = 0; v < 8; ++v) acc2[nt][v] = bv;
    }
    const v16h* pk2 = (const v16h*)pkw2;
#pragma unroll
    for (int ks = 0; ks < 4; ++ks) {
        int kb = ks * 32 + half * 8;
        v8h lo = *(const v8h*)(hsh + mrow * 128 + kb);
        v8h hi = *(const v8h*)(hsh + mrow * 128 + kb + 16);
        v16h a = cat8(lo, hi);
#pragma unroll
        for (int nt = 0; nt < 4; ++nt)
            acc2[nt] = wmma_f16(a, pk2[(ks * 4 + nt) * 32 + lane], acc2[nt]);
    }
#pragma unroll
    for (int nt = 0; nt < 4; ++nt)
#pragma unroll
        for (int v = 0; v < 8; ++v) {
            size_t idx = (size_t)(rowbase + half * 8 + v) * 64 + nt * 16 + mrow;
            xout[idx] = acc2[nt][v];
        }
}

// ---------------- graph norm + ELU (contiguous 200-node graphs; thread owns one channel)
__global__ void graphnorm_elu(float* __restrict__ x, _Float16* __restrict__ xh,
                              const float* __restrict__ gw, const float* __restrict__ gb,
                              const float* __restrict__ ga) {
    int g = blockIdx.x, c = threadIdx.x;
    float* base = x + (size_t)g * NPG * 64;
    _Float16* hb = xh + (size_t)g * NPG * 64;
    float s = 0.f;
    for (int j = 0; j < NPG; ++j) s += base[j * 64 + c];
    float mean = s * (1.f / NPG);
    float ma = mean * ga[c];
    float vs = 0.f;
    for (int j = 0; j < NPG; ++j) { float d = base[j * 64 + c] - ma; vs += d * d; }
    float inv = rsqrtf(vs * (1.f / NPG) + 1e-5f);
    float w = gw[c], bb = gb[c];
    for (int j = 0; j < NPG; ++j) {
        float d = base[j * 64 + c] - ma;
        float o = d * inv * w + bb;
        o = o > 0.f ? o : (expf(o) - 1.f);
        base[j * 64 + c] = o;
        hb[j * 64 + c] = (_Float16)o;
    }
}

// ---------------- attention gate: gate = elu(x @ W1 + b1) @ W2 + b2
__global__ void gate_kernel(const float* __restrict__ x, const float* __restrict__ W1,
                            const float* __restrict__ b1, const float* __restrict__ W2,
                            const float* __restrict__ b2, float* __restrict__ gate, int N) {
    int n = blockIdx.x * blockDim.x + threadIdx.x;
    if (n >= N) return;
    float xv[64];
#pragma unroll
    for (int c = 0; c < 64; ++c) xv[c] = x[(size_t)n * 64 + c];
    float gacc = b2[0];
    for (int k = 0; k < 32; ++k) {
        float s = b1[k];
#pragma unroll
        for (int c = 0; c < 64; ++c) s += xv[c] * W1[c * 32 + k];
        s = s > 0.f ? s : (expf(s) - 1.f);
        gacc += s * W2[k];
    }
    gate[n] = gacc;
}

// ---------------- per-graph softmax pooling + classifier
__global__ void pool_cls(const float* __restrict__ x, const float* __restrict__ gate,
                         const float* __restrict__ clsW, const float* __restrict__ clsb,
                         float* __restrict__ out) {
    __shared__ float red[64];
    __shared__ float ebuf[NPG];
    __shared__ float psh[64];
    int g = blockIdx.x, t = threadIdx.x;  // 64 threads
    const float* gb = gate + (size_t)g * NPG;
    float m = -1e30f;
    for (int j = t; j < NPG; j += 64) m = fmaxf(m, gb[j]);
    red[t] = m; __syncthreads();
    for (int s = 32; s > 0; s >>= 1) { if (t < s) red[t] = fmaxf(red[t], red[t + s]); __syncthreads(); }
    float gmax = red[0]; __syncthreads();
    float sum = 0.f;
    for (int j = t; j < NPG; j += 64) { float e = expf(gb[j] - gmax); ebuf[j] = e; sum += e; }
    red[t] = sum; __syncthreads();
    for (int s = 32; s > 0; s >>= 1) { if (t < s) red[t] += red[t + s]; __syncthreads(); }
    float denom = red[0]; __syncthreads();
    const float* xb = x + (size_t)g * NPG * 64;
    float p = 0.f;
    for (int j = 0; j < NPG; ++j) p += xb[j * 64 + t] * ebuf[j];
    psh[t] = p / denom; __syncthreads();
    if (t < 2) {
        float o = clsb[t];
        for (int c = 0; c < 64; ++c) o += psh[c] * clsW[c * 2 + t];
        out[g * 2 + t] = o;
    }
}

extern "C" void kernel_launch(void* const* d_in, const int* in_sizes, int n_in,
                              void* d_out, int out_size, void* d_ws, size_t ws_size,
                              hipStream_t stream) {
    const int*   x_idx  = (const int*)  d_in[0];
    const int*   eidx   = (const int*)  d_in[1];
    const float* eattr  = (const float*)d_in[2];
    const int*   batch  = (const int*)  d_in[3];
    const float* demog  = (const float*)d_in[4];
    const float* emb    = (const float*)d_in[5];
    const float* projW  = (const float*)d_in[6];
    const float* projB  = (const float*)d_in[7];
    const float* relW   = (const float*)d_in[8];
    const float* relB   = (const float*)d_in[9];
    const float* rootW  = (const float*)d_in[10];
    const float* dW0    = (const float*)d_in[11];
    const float* dB0    = (const float*)d_in[12];
    const float* dW1    = (const float*)d_in[13];
    const float* dB1    = (const float*)d_in[14];
    const float* dnW1   = (const float*)d_in[15];
    const float* dnB1   = (const float*)d_in[16];
    const float* dnW2   = (const float*)d_in[17];
    const float* dnB2   = (const float*)d_in[18];
    const float* gnw    = (const float*)d_in[19];
    const float* gnb    = (const float*)d_in[20];
    const float* gna    = (const float*)d_in[21];
    const float* attW1  = (const float*)d_in[22];
    const float* attB1  = (const float*)d_in[23];
    const float* attW2  = (const float*)d_in[24];
    const float* attB2  = (const float*)d_in[25];
    const float* clsW   = (const float*)d_in[26];
    const float* clsB   = (const float*)d_in[27];

    const int N = in_sizes[0];
    const int E = in_sizes[1] / 2;
    const int G = in_sizes[4] / 4;
    const int* src = eidx;        // edge_index row 0
    const int* dst = eidx + E;    // edge_index row 1

    char* w = (char*)d_ws;
    float* X    = (float*)w;    w += (size_t)N * 64 * 4;
    float* Y    = (float*)w;    w += (size_t)N * 64 * 4;
    float* AGG  = (float*)w;    w += (size_t)N * 64 * 4;
    _Float16* XH   = (_Float16*)w; w += (size_t)N * 64 * 2;
    _Float16* YH   = (_Float16*)w; w += (size_t)N * 64 * 2;
    _Float16* AGGH = (_Float16*)w; w += (size_t)N * 64 * 2;
    float* DA   = (float*)w;    w += (size_t)G * 16 * 4;
    float* DB   = (float*)w;    w += (size_t)G * 16 * 4;
    float* GATE = (float*)w;    w += (size_t)N * 4;
    _Float16* PKW = (_Float16*)w;  // 65536 f16 of packed fragments

    _Float16* PK_REL[3], *PK_ROOT[3], *PK_DW1[2], *PK_DW2[2];
    for (int i = 0; i < 3; ++i) PK_REL[i]  = PKW + (size_t)i * 4096;
    for (int i = 0; i < 3; ++i) PK_ROOT[i] = PKW + 12288 + (size_t)i * 4096;
    for (int i = 0; i < 2; ++i) PK_DW1[i]  = PKW + 24576 + (size_t)i * 12288;
    for (int i = 0; i < 2; ++i) PK_DW2[i]  = PKW + 49152 + (size_t)i * 8192;

    // ---- pack weights into WMMA fragment order (f16), once per launch
    for (int i = 0; i < 3; ++i) {
        pack_weights<<<16, 256, 0, stream>>>(relW  + (size_t)i * 64 * 64, PK_REL[i],  64, 64, 2, 4);
        pack_weights<<<16, 256, 0, stream>>>(rootW + (size_t)i * 64 * 64, PK_ROOT[i], 64, 64, 2, 4);
    }
    for (int i = 0; i < 2; ++i) {
        pack_weights<<<48, 256, 0, stream>>>(dnW1 + (size_t)i * 80 * 128, PK_DW1[i], 80, 128, 3, 8);
        pack_weights<<<32, 256, 0, stream>>>(dnW2 + (size_t)i * 128 * 64, PK_DW2[i], 128, 64, 4, 4);
    }

    embed_proj<<<(N * 64 + 255) / 256, 256, 0, stream>>>(x_idx, emb, projW, projB, X, XH, N);

    for (int i = 0; i < 3; ++i) {
        (void)hipMemsetAsync(AGG, 0, (size_t)N * 64 * sizeof(float), stream);
        edge_scatter<<<((size_t)E * 64 + 255) / 256, 256, 0, stream>>>(src, dst, eattr, X, AGG, E);
        f32_to_f16<<<(N * 16 + 255) / 256, 256, 0, stream>>>(AGG, AGGH, N * 16);
        conv_wmma<<<N / 16, 32, 0, stream>>>(AGGH, XH, PK_REL[i], relB + i * 64,
                                             PK_ROOT[i], Y, YH);
        if (i < 2) {
            const float* din  = (i == 0) ? demog : DA;
            float*       dout = (i == 0) ? DA    : DB;
            const float* dW   = (i == 0) ? dW0   : dW1;
            const float* dB   = (i == 0) ? dB0   : dB1;
            int          K    = (i == 0) ? 4     : 16;
            demo_mlp<<<(G * 16 + 255) / 256, 256, 0, stream>>>(din, dW, dB, dout, K, G);
            down_wmma<<<N / 16, 32, 0, stream>>>(YH, dout, batch, PK_DW1[i], dnB1 + i * 128,
                                                 PK_DW2[i], dnB2 + i * 64, X);
            graphnorm_elu<<<G, 64, 0, stream>>>(X, XH, gnw + i * 64, gnb + i * 64, gna + i * 64);
        }
    }

    gate_kernel<<<(N + 255) / 256, 256, 0, stream>>>(Y, attW1, attB1, attW2, attB2, GATE, N);
    pool_cls<<<G, 64, 0, stream>>>(Y, GATE, clsW, clsB, (float*)d_out);
}
